// AttentionLayer_71863392796954
// MI455X (gfx1250) — compile-verified, matrix operand order checked by reading, not run on hardware
//
#include <hip/hip_runtime.h>

// ---------------------------------------------------------------------------
// CDNA5 (gfx1250) WMMA attention layer.
// Shapes (from reference): B=16 T=12 N=1024 D=512 H=8 HD=64 C=128, pool group G=8.
// All matrix math lowered to v_wmma_f32_16x16x32_bf16 (fp32 accumulate).
// ---------------------------------------------------------------------------

typedef __attribute__((ext_vector_type(16))) __bf16 v16bf;
typedef __attribute__((ext_vector_type(8)))  __bf16 v8bf;
typedef __attribute__((ext_vector_type(8)))  float  v8f;

union Frag16 { v16bf f; v8bf h[2]; };

constexpr int kB = 16, kT = 12, kN = 1024, kD = 512;
constexpr int kH = 8, kHD = 64, kC = 128, kG = 8;

__device__ __forceinline__ unsigned short f2bf(float f) {
  unsigned int u = __builtin_bit_cast(unsigned int, f);
  u += 0x7FFFu + ((u >> 16) & 1u);            // round-to-nearest-even
  return (unsigned short)(u >> 16);
}

__device__ __forceinline__ v8f wmma_bf16(const Frag16& a, const Frag16& b, v8f c) {
  // D = A(16x32) * B(32x16) + C(16x16), f32 accumulate
  return __builtin_amdgcn_wmma_f32_16x16x32_bf16(false, a.f, false, b.f,
                                                 (short)0, c, false, false);
}

// ---------------------------------------------------------------------------
// Mean pool over node groups of 8:  xp[bt,c,d] = mean_g x[bt, c*8+g, d]
// Pure bandwidth kernel.
// ---------------------------------------------------------------------------
__global__ void pool_kernel(const float* __restrict__ x, float* __restrict__ xp,
                            int total) {
  int i = blockIdx.x * blockDim.x + threadIdx.x;
  if (i >= total) return;
  int d    = i & (kD - 1);
  int rest = i / kD;
  int c    = rest & (kC - 1);
  int bt   = rest / kC;
  const float* src = x + ((size_t)bt * kN + (size_t)c * kG) * kD + d;
  float s = 0.f;
#pragma unroll
  for (int g = 0; g < kG; ++g) s += src[(size_t)g * kD];
  xp[i] = s * (1.0f / kG);
}

// ---------------------------------------------------------------------------
// Tiled GEMM: Out[M,Nout] = A[M,K] * W[K,Nout] + bias.
// Block tile 128x128, K-step 32, 8 waves each own a 32x64 sub-tile.
// A converted to bf16 into LDS; W tile stored transposed ([n][k]) so the
// B-operand fragment load pattern matches the A-operand pattern
// (B is consumed column-major per the CDNA5 WMMA spec).
// ---------------------------------------------------------------------------
constexpr int BM = 128, BN = 128, BK = 32;

__device__ __forceinline__ void load_a_tile(const float* A, unsigned short (*ls)[BK],
                                            int tid, int bm0, int k0, int K) {
  int col = (tid & 7) * 4;
  int row = tid >> 3;                               // 0..31
#pragma unroll
  for (int rr = 0; rr < BM; rr += 32) {
    float4 v = *reinterpret_cast<const float4*>(&A[(size_t)(bm0 + row + rr) * K + k0 + col]);
    ls[row + rr][col + 0] = f2bf(v.x);
    ls[row + rr][col + 1] = f2bf(v.y);
    ls[row + rr][col + 2] = f2bf(v.z);
    ls[row + rr][col + 3] = f2bf(v.w);
  }
}

__device__ __forceinline__ void load_a_tile(const unsigned short* A, unsigned short (*ls)[BK],
                                            int tid, int bm0, int k0, int K) {
  int col = (tid & 3) * 8;
  int row = tid >> 2;                               // 0..63
#pragma unroll
  for (int rr = 0; rr < BM; rr += 64) {
    uint4 v = *reinterpret_cast<const uint4*>(&A[(size_t)(bm0 + row + rr) * K + k0 + col]);
    *reinterpret_cast<uint4*>(&ls[row + rr][col]) = v;
  }
}

__device__ __forceinline__ void store_elem(float* C, size_t idx, float v) { C[idx] = v; }
__device__ __forceinline__ void store_elem(unsigned short* C, size_t idx, float v) { C[idx] = f2bf(v); }

template <typename AT, typename OT>
__global__ void __launch_bounds__(256)
gemm_bias_kernel(const AT* __restrict__ A, const float* __restrict__ W,
                 const float* __restrict__ bias, OT* __restrict__ Out,
                 int M, int K, int Nout) {
  __shared__ unsigned short lsA[2][BM][BK];   // 8 KB x2
  __shared__ unsigned short lsB[2][BN][BK];   // 8 KB x2, transposed W tile [n][k]

  const int tid  = threadIdx.x;
  const int lane = tid & 31;
  const int wave = tid >> 5;
  const int wm   = (wave & 3) * 32;
  const int wn   = (wave >> 2) * 64;
  const int bm0  = blockIdx.x * BM;
  const int bn0  = blockIdx.y * BN;
  const int lm   = lane & 15;                 // A: M row / B: N col within 16-tile
  const int kh   = lane >> 4;                 // K-half select per ISA A/B layout

  v8f acc[2][4] = {};

  auto loadB = [&](int buf, int k0) {
    int kk = tid >> 5;                        // 0..7
    int nn = (tid & 31) * 4;                  // 0..124
#pragma unroll
    for (int kr = 0; kr < BK; kr += 8) {
      float4 v = *reinterpret_cast<const float4*>(&W[(size_t)(k0 + kk + kr) * Nout + bn0 + nn]);
      lsB[buf][nn + 0][kk + kr] = f2bf(v.x);
      lsB[buf][nn + 1][kk + kr] = f2bf(v.y);
      lsB[buf][nn + 2][kk + kr] = f2bf(v.z);
      lsB[buf][nn + 3][kk + kr] = f2bf(v.w);
    }
  };

  load_a_tile(A, lsA[0], tid, bm0, 0, K);
  loadB(0, 0);
  __syncthreads();

  const int nk = K / BK;
  for (int ks = 0; ks < nk; ++ks) {
    const int buf = ks & 1;
    if (ks + 1 < nk) {                        // prefetch next K-slab (double buffer)
      load_a_tile(A, lsA[buf ^ 1], tid, bm0, (ks + 1) * BK, K);
      loadB(buf ^ 1, (ks + 1) * BK);
    }
    Frag16 af[2], bfr[4];
#pragma unroll
    for (int i = 0; i < 2; ++i) {
      const unsigned short* p = &lsA[buf][wm + i * 16 + lm][kh * 8];
      af[i].h[0] = *reinterpret_cast<const v8bf*>(p);
      af[i].h[1] = *reinterpret_cast<const v8bf*>(p + 16);
    }
#pragma unroll
    for (int j = 0; j < 4; ++j) {
      const unsigned short* p = &lsB[buf][wn + j * 16 + lm][kh * 8];
      bfr[j].h[0] = *reinterpret_cast<const v8bf*>(p);
      bfr[j].h[1] = *reinterpret_cast<const v8bf*>(p + 16);
    }
#pragma unroll
    for (int i = 0; i < 2; ++i)
#pragma unroll
      for (int j = 0; j < 4; ++j)
        acc[i][j] = wmma_bf16(af[i], bfr[j], acc[i][j]);
    __syncthreads();
  }

  // Epilogue: C-matrix layout -> lane holds col (lane&15), rows r + 8*(lane>>4)
#pragma unroll
  for (int i = 0; i < 2; ++i) {
    const int rbase = bm0 + wm + i * 16 + (kh << 3);
#pragma unroll
    for (int j = 0; j < 4; ++j) {
      const int col = bn0 + wn + j * 16 + lm;
      const float bv = bias[col];
#pragma unroll
      for (int r = 0; r < 8; ++r)
        store_elem(Out, (size_t)(rbase + r) * Nout + col, acc[i][j][r] + bv);
    }
  }
}

// ---------------------------------------------------------------------------
// Attention: per (b,t,h) and a 128-row slab of N.
//   S = q(128x64) . k^T(64x128) * 0.125 + adp_pos   (via 2 WMMAs per 16x16 tile)
//   P = softmax_rows(S)                              (shfl_xor within 16-lane half)
//   O = P(128x128) . v(128x64)                       (P staged in LDS, v^T in LDS)
// ---------------------------------------------------------------------------
__global__ void __launch_bounds__(256)
attn_kernel(const unsigned short* __restrict__ q, const unsigned short* __restrict__ kmat,
            const unsigned short* __restrict__ vmat, const float* __restrict__ adp_pos,
            unsigned short* __restrict__ attn_out) {
  __shared__ unsigned short lds_p[128][kC];    // probabilities, 32 KB
  __shared__ unsigned short lds_vt[kHD][kC];   // v transposed [hd][c], 16 KB

  const int tid  = threadIdx.x;
  const int lane = tid & 31;
  const int wave = tid >> 5;                   // 0..7, 16 rows each
  const int bth  = blockIdx.x;
  const int h    = bth & (kH - 1);
  const int bt   = bth / kH;
  const int row0 = blockIdx.y * 128 + wave * 16;
  const int lm   = lane & 15;
  const int kh   = lane >> 4;

  // Stage v^T (B operand for P.V must be column-major of V = [hd][c]).
  const size_t vbase = (size_t)bt * kC * kD + (size_t)h * kHD;
  for (int i = tid; i < kHD * kC; i += 256) {
    int hd = i >> 7;             // / kC
    int c  = i & (kC - 1);
    lds_vt[hd][c] = vmat[vbase + (size_t)c * kD + hd];
  }
  __syncthreads();

  // q fragments straight from global (A layout: contiguous 8-element halves).
  Frag16 qa[2];
  const size_t qrow = ((size_t)bt * kN + row0 + lm) * kD + (size_t)h * kHD;
#pragma unroll
  for (int kf = 0; kf < 2; ++kf) {
    qa[kf].h[0] = *reinterpret_cast<const v8bf*>(&q[qrow + kf * 32 + kh * 8]);
    qa[kf].h[1] = *reinterpret_cast<const v8bf*>(&q[qrow + kf * 32 + kh * 8 + 16]);
  }

  // Scores: 8 accumulators cover C=128.
  v8f sc[8] = {};
#pragma unroll
  for (int cf = 0; cf < 8; ++cf) {
    const size_t krow = ((size_t)bt * kC + cf * 16 + lm) * kD + (size_t)h * kHD;
#pragma unroll
    for (int kf = 0; kf < 2; ++kf) {
      Frag16 kb;  // k rows are B columns (k^T consumed column-major == k row-major)
      kb.h[0] = *reinterpret_cast<const v8bf*>(&kmat[krow + kf * 32 + kh * 8]);
      kb.h[1] = *reinterpret_cast<const v8bf*>(&kmat[krow + kf * 32 + kh * 8 + 16]);
      sc[cf] = wmma_bf16(qa[kf], kb, sc[cf]);
    }
  }

  // Softmax per row (value at (r, lane&15) per frag; reduce across 16 lanes).
  const float scale = 0.125f;                  // 1/sqrt(HD)
#pragma unroll
  for (int r = 0; r < 8; ++r) {
    const int row = row0 + r + (kh << 3);
    float cv[8];
    float mx = -3.0e38f;
#pragma unroll
    for (int cf = 0; cf < 8; ++cf) {
      cv[cf] = sc[cf][r] * scale + adp_pos[(size_t)row * kC + cf * 16 + lm];
      mx = fmaxf(mx, cv[cf]);
    }
#pragma unroll
    for (int off = 1; off < 16; off <<= 1)
      mx = fmaxf(mx, __shfl_xor(mx, off, 32));
    float sum = 0.f;
#pragma unroll
    for (int cf = 0; cf < 8; ++cf) {
      cv[cf] = __expf(cv[cf] - mx);
      sum += cv[cf];
    }
#pragma unroll
    for (int off = 1; off < 16; off <<= 1)
      sum += __shfl_xor(sum, off, 32);
    const float inv = 1.0f / sum;
#pragma unroll
    for (int cf = 0; cf < 8; ++cf)
      lds_p[wave * 16 + r + (kh << 3)][cf * 16 + lm] = f2bf(cv[cf] * inv);
  }

  // O = P . V  (K=128 -> 4 WMMA K-steps; HD=64 -> 4 output tiles).
  v8f oacc[4] = {};
#pragma unroll
  for (int ks = 0; ks < 4; ++ks) {
    Frag16 pa;
    const unsigned short* pp = &lds_p[wave * 16 + lm][ks * 32 + kh * 8];
    pa.h[0] = *reinterpret_cast<const v8bf*>(pp);
    pa.h[1] = *reinterpret_cast<const v8bf*>(pp + 16);
#pragma unroll
    for (int nf = 0; nf < 4; ++nf) {
      Frag16 vb;
      const unsigned short* vp = &lds_vt[nf * 16 + lm][ks * 32 + kh * 8];
      vb.h[0] = *reinterpret_cast<const v8bf*>(vp);
      vb.h[1] = *reinterpret_cast<const v8bf*>(vp + 16);
      oacc[nf] = wmma_bf16(pa, vb, oacc[nf]);
    }
  }

#pragma unroll
  for (int nf = 0; nf < 4; ++nf) {
    const int col = h * kHD + nf * 16 + lm;
#pragma unroll
    for (int r = 0; r < 8; ++r) {
      const int row = row0 + r + (kh << 3);
      attn_out[((size_t)bt * kN + row) * kD + col] = f2bf(oacc[nf][r]);
    }
  }
}

// ---------------------------------------------------------------------------
extern "C" void kernel_launch(void* const* d_in, const int* in_sizes, int n_in,
                              void* d_out, int out_size, void* d_ws, size_t ws_size,
                              hipStream_t stream) {
  const float* x   = (const float*)d_in[0];
  const float* Wq  = (const float*)d_in[1];
  const float* bq  = (const float*)d_in[2];
  const float* Wk  = (const float*)d_in[3];
  const float* bk  = (const float*)d_in[4];
  const float* Wv  = (const float*)d_in[5];
  const float* bv  = (const float*)d_in[6];
  const float* Wo  = (const float*)d_in[7];
  const float* bo  = (const float*)d_in[8];
  const float* adp = (const float*)d_in[9];
  float* out = (float*)d_out;

  const size_t qElems = (size_t)kB * kT * kN * kD;   // 100,663,296
  const size_t kElems = (size_t)kB * kT * kC * kD;   //  12,582,912

  unsigned short* qb  = (unsigned short*)d_ws;       // q  bf16
  unsigned short* kb_ = qb + qElems;                 // k  bf16 (pooled)
  unsigned short* vb_ = kb_ + kElems;                // v  bf16 (pooled)
  unsigned short* ab  = vb_ + kElems;                // attn output bf16
  float* xp = (float*)ab;                            // pooled x, dead before ab is written

  // 1) pool(x) -> xp   (linearity: pool before k/v projections, 8x fewer GEMM FLOPs)
  {
    int total = (int)kElems;
    pool_kernel<<<dim3((total + 255) / 256), 256, 0, stream>>>(x, xp, total);
  }
  // 2) q = x @ Wq + bq  -> bf16
  {
    dim3 g((kB * kT * kN) / BM, kD / BN);
    gemm_bias_kernel<float, unsigned short><<<g, 256, 0, stream>>>(x, Wq, bq, qb, kB * kT * kN, kD, kD);
  }
  // 3,4) k = xp @ Wk + bk ; v = xp @ Wv + bv  -> bf16
  {
    dim3 g((kB * kT * kC) / BM, kD / BN);
    gemm_bias_kernel<float, unsigned short><<<g, 256, 0, stream>>>(xp, Wk, bk, kb_, kB * kT * kC, kD, kD);
    gemm_bias_kernel<float, unsigned short><<<g, 256, 0, stream>>>(xp, Wv, bv, vb_, kB * kT * kC, kD, kD);
  }
  // 5) attention -> ab (bf16, merged-head layout (B,T,N,D))
  {
    dim3 g(kB * kT * kH, kN / 128);
    attn_kernel<<<g, 256, 0, stream>>>(qb, kb_, vb_, adp, ab);
  }
  // 6) out = ab @ Wo + bo -> fp32
  {
    dim3 g((kB * kT * kN) / BM, kD / BN);
    gemm_bias_kernel<unsigned short, float><<<g, 256, 0, stream>>>(ab, Wo, bo, out, kB * kT * kN, kD, kD);
  }
  (void)in_sizes; (void)n_in; (void)out_size; (void)ws_size;
}
